// SDEDecoder_66675072303316
// MI455X (gfx1250) — compile-verified, matrix operand order checked by reading, not run on hardware
//
#include <hip/hip_runtime.h>
#include <hip/hip_bf16.h>
#include <math.h>

#define MDIM 6
#define ADIM 4096
#define HDIM 128
#define TDIM 30
#define NDIM (MDIM * ADIM)   // 24576
#define HIST 50
#define DT_F 0.2f
#define SQDT_F 0.4472135954999579f
#define EPSLN 1e-5f
#define ACT_STRIDE 132       // bf16 elements per row (padded, 4B aligned, bank-friendly)

typedef __attribute__((ext_vector_type(16))) __bf16 v16bf;
typedef __attribute__((ext_vector_type(8)))  float  v8f;

union FragU { v16bf v; unsigned int u[8]; };

// ---- fast transcendentals (gfx1250 has hardware v_tanh_f32 / v_exp_f32) ----
#if __has_builtin(__builtin_amdgcn_tanhf)
#define FAST_TANH(x) __builtin_amdgcn_tanhf(x)
#elif __has_builtin(__builtin_amdgcn_tanh_f32)
#define FAST_TANH(x) __builtin_amdgcn_tanh_f32(x)
#else
static __device__ __forceinline__ float fast_tanh_asm(float x) {
  float r;
  asm("v_tanh_f32 %0, %1\n\tv_nop" : "=v"(r) : "v"(x));  // v_nop covers TRANS RAW hazard
  return r;
}
#define FAST_TANH(x) fast_tanh_asm(x)
#endif

#if __has_builtin(__builtin_amdgcn_rcpf)
#define FAST_RCP(x) __builtin_amdgcn_rcpf(x)
#else
#define FAST_RCP(x) (1.0f / (x))
#endif

#define FAST_EXP(x) __expf(x)

struct ScanLds {
  // transposed bf16 weights: W[i][n*128 + k] = w[k][n]
  unsigned short W[7][HDIM * HDIM];          // 229376 B
  unsigned short actY[HDIM * ACT_STRIDE];    // 33792 B  (bf16 copy of y)
  unsigned short actH[HDIM * ACT_STRIDE];    // 33792 B  (layer intermediates)
  float fb1[HDIM], fts[HDIM], ftc[HDIM], fb2[HDIM], fb3[HDIM];
  float gb1[HDIM], gts[HDIM], gtc[HDIM], gb2[HDIM], gw3[HDIM];
  float db1[HDIM], dgm[HDIM], dbe[HDIM];
  float sb1[HDIM], sgm[HDIM], sbe[HDIM];
  float ab[HDIM],  ag[HDIM],  abe[HDIM];
  float dw2[2 * HDIM], sw2[2 * HDIM];
  float gb3s, db2s[2], sb2s[2];
};

__device__ __forceinline__ unsigned short f2bf(float f) {
  unsigned int u = __float_as_uint(f);
  u += 0x7FFFu + ((u >> 16) & 1u);           // round-to-nearest-even
  return (unsigned short)(u >> 16);
}

__device__ __forceinline__ float red16(float v) {
  v += __shfl_xor(v, 1, 32);
  v += __shfl_xor(v, 2, 32);
  v += __shfl_xor(v, 4, 32);
  v += __shfl_xor(v, 8, 32);
  return v;
}

__device__ void load_wT(unsigned short* dst, const float* __restrict__ src, int tid) {
  for (int i = tid; i < HDIM * HDIM; i += 256) {
    int k = i >> 7, n = i & 127;
    dst[n * HDIM + k] = f2bf(src[i]);
  }
}

__device__ void load_vec(float* dst, const float* __restrict__ src, int n, int tid) {
  for (int i = tid; i < n; i += 256) dst[i] = src[i];
}

// acc += act[r0..r0+15, 0:128] @ W (W stored transposed [n][k] bf16)
// B fragments double-buffered (stays under the 256-VGPR boundary).
__device__ void gemm_acc(const unsigned short* __restrict__ act,
                         const unsigned short* __restrict__ Wt,
                         int r0, int lane, v8f acc[8]) {
  const int colB = lane & 15;
  const int khi  = (lane >> 4) << 3;          // +8 for lanes 16..31
  const unsigned short* arow = act + (r0 + colB) * ACT_STRIDE;
#pragma unroll
  for (int k0 = 0; k0 < HDIM; k0 += 32) {
    FragU af;
#pragma unroll
    for (int p = 0; p < 8; ++p) {
      int k = k0 + ((p & 4) << 2) + khi + ((p & 3) << 1);
      af.u[p] = *(const unsigned int*)(arow + k);
    }
    FragU b0, b1;
#pragma unroll
    for (int p = 0; p < 8; ++p) {
      int k = k0 + ((p & 4) << 2) + khi + ((p & 3) << 1);
      b0.u[p] = *(const unsigned int*)(Wt + colB * HDIM + k);
    }
#pragma unroll
    for (int c = 0; c < 8; ++c) {
      FragU* cur = (c & 1) ? &b1 : &b0;
      FragU* nxt = (c & 1) ? &b0 : &b1;
      if (c < 7) {
        const unsigned short* brow = Wt + ((c + 1) * 16 + colB) * HDIM;
#pragma unroll
        for (int p = 0; p < 8; ++p) {
          int k = k0 + ((p & 4) << 2) + khi + ((p & 3) << 1);
          nxt->u[p] = *(const unsigned int*)(brow + k);
        }
      }
      acc[c] = __builtin_amdgcn_wmma_f32_16x16x32_bf16(
          false, af.v, false, cur->v, (short)0, acc[c], false, false);
    }
  }
}

__device__ __forceinline__ void zero_acc(v8f acc[8]) {
#pragma unroll
  for (int c = 0; c < 8; ++c)
#pragma unroll
    for (int r = 0; r < 8; ++r) acc[c][r] = 0.0f;
}

// D-layout (row = r0 + r + hi*8, col = c*16 + colB) -> bf16 LDS buffer
__device__ void store_act_bf(unsigned short* buf, int r0, int lane, const v8f acc[8]) {
  const int rB = r0 + ((lane >> 4) << 3);
  const int colB = lane & 15;
#pragma unroll
  for (int c = 0; c < 8; ++c)
#pragma unroll
    for (int r = 0; r < 8; ++r)
      buf[(rB + r) * ACT_STRIDE + c * 16 + colB] = f2bf(acc[c][r]);
}

// bias already added; LayerNorm over cols (per row) then ReLU
__device__ void ln_relu(v8f acc[8], int colB, const float* g, const float* be) {
  float mu[8], rs[8];
#pragma unroll
  for (int r = 0; r < 8; ++r) {
    float s = 0.f;
#pragma unroll
    for (int c = 0; c < 8; ++c) s += acc[c][r];
    mu[r] = red16(s) * (1.0f / 128.0f);
  }
#pragma unroll
  for (int r = 0; r < 8; ++r) {
    float s = 0.f;
#pragma unroll
    for (int c = 0; c < 8; ++c) { float d = acc[c][r] - mu[r]; s += d * d; }
    rs[r] = rsqrtf(red16(s) * (1.0f / 128.0f) + EPSLN);
  }
#pragma unroll
  for (int c = 0; c < 8; ++c)
#pragma unroll
    for (int r = 0; r < 8; ++r) {
      int col = c * 16 + colB;
      float v = (acc[c][r] - mu[r]) * rs[r] * g[col] + be[col];
      acc[c][r] = v > 0.f ? v : 0.f;
    }
}

__global__ __launch_bounds__(256) void sde_scan_kernel(
    const float* __restrict__ local_embed, const float* __restrict__ global_embed,
    const float* __restrict__ noise,
    const float* __restrict__ aggr_w, const float* __restrict__ aggr_b,
    const float* __restrict__ aggr_g, const float* __restrict__ aggr_be,
    const float* __restrict__ f_w1, const float* __restrict__ f_b1,
    const float* __restrict__ f_w2, const float* __restrict__ f_b2,
    const float* __restrict__ f_w3, const float* __restrict__ f_b3,
    const float* __restrict__ g_w1, const float* __restrict__ g_b1,
    const float* __restrict__ g_w2, const float* __restrict__ g_b2,
    const float* __restrict__ g_w3, const float* __restrict__ g_b3,
    const float* __restrict__ d_w1, const float* __restrict__ d_b1,
    const float* __restrict__ d_g,  const float* __restrict__ d_be,
    const float* __restrict__ d_w2, const float* __restrict__ d_b2,
    const float* __restrict__ s_w1, const float* __restrict__ s_b1,
    const float* __restrict__ s_g,  const float* __restrict__ s_be,
    const float* __restrict__ s_w2, const float* __restrict__ s_b2,
    float* __restrict__ out_loc) {
  extern __shared__ __align__(16) char smem_raw[];
  ScanLds& S = *reinterpret_cast<ScanLds*>(smem_raw);

  const int tid  = threadIdx.x;
  const int lane = tid & 31;
  const int wave = tid >> 5;
  const int r0   = wave * 16;
  const int colB = lane & 15;
  const int rHi  = (lane >> 4) << 3;
  const int gBase = blockIdx.x * 128;

  // ---- prologue: weights -> LDS (bf16 transposed), small vectors ----
  load_wT(S.W[0], f_w1, tid);                 // rows 0..127 of [130,128]
  load_wT(S.W[1], f_w2, tid);
  load_wT(S.W[2], f_w3, tid);
  load_wT(S.W[3], g_w1, tid);
  load_wT(S.W[4], g_w2, tid);
  load_wT(S.W[5], aggr_w, tid);               // aggr rows 0..127 (global part)
  load_wT(S.W[6], aggr_w + HDIM * HDIM, tid); // aggr rows 128..255 (local part)
  load_vec(S.fb1, f_b1, HDIM, tid);
  load_vec(S.fts, f_w1 + 128 * HDIM, HDIM, tid);
  load_vec(S.ftc, f_w1 + 129 * HDIM, HDIM, tid);
  load_vec(S.fb2, f_b2, HDIM, tid);
  load_vec(S.fb3, f_b3, HDIM, tid);
  load_vec(S.gb1, g_b1, HDIM, tid);
  load_vec(S.gts, g_w1 + 128 * HDIM, HDIM, tid);
  load_vec(S.gtc, g_w1 + 129 * HDIM, HDIM, tid);
  load_vec(S.gb2, g_b2, HDIM, tid);
  load_vec(S.gw3, g_w3, HDIM, tid);
  load_vec(S.ab, aggr_b, HDIM, tid);
  load_vec(S.ag, aggr_g, HDIM, tid);
  load_vec(S.abe, aggr_be, HDIM, tid);
  if (tid == 0) {
    S.gb3s = g_b3[0];
    S.db2s[0] = d_b2[0]; S.db2s[1] = d_b2[1];
    S.sb2s[0] = s_b2[0]; S.sb2s[1] = s_b2[1];
  }
  // input rows -> bf16 act buffers: actY = global_embed rows, actH = local rows
  for (int i = tid; i < 128 * HDIM; i += 256) {
    int row = i >> 7, k = i & 127;
    int g = gBase + row;
    int a = g & (ADIM - 1);
    S.actY[row * ACT_STRIDE + k] = f2bf(global_embed[(size_t)g * HDIM + k]);
    S.actH[row * ACT_STRIDE + k] = f2bf(local_embed[(size_t)a * HDIM + k]);
  }
  __syncthreads();

  // ---- h0 = relu(LN(cat(global, local) @ aggr_w + aggr_b)) ----
  v8f acc[8];
  zero_acc(acc);
  gemm_acc(S.actY, S.W[5], r0, lane, acc);
  gemm_acc(S.actH, S.W[6], r0, lane, acc);
#pragma unroll
  for (int c = 0; c < 8; ++c)
#pragma unroll
    for (int r = 0; r < 8; ++r) acc[c][r] += S.ab[c * 16 + colB];
  ln_relu(acc, colB, S.ag, S.abe);
  v8f y[8];
#pragma unroll
  for (int c = 0; c < 8; ++c) y[c] = acc[c];
  __syncthreads();                             // all done reading W5/W6
  store_act_bf(S.actY, r0, lane, y);
  load_wT(S.W[5], d_w1, tid);                  // reuse slots for decoder heads
  load_wT(S.W[6], s_w1, tid);
  load_vec(S.db1, d_b1, HDIM, tid);
  load_vec(S.dgm, d_g, HDIM, tid);
  load_vec(S.dbe, d_be, HDIM, tid);
  load_vec(S.dw2, d_w2, 2 * HDIM, tid);
  load_vec(S.sb1, s_b1, HDIM, tid);
  load_vec(S.sgm, s_g, HDIM, tid);
  load_vec(S.sbe, s_be, HDIM, tid);
  load_vec(S.sw2, s_w2, 2 * HDIM, tid);
  __syncthreads();

  // ---- Euler–Maruyama scan ----
  // NOTE: no barriers inside the loop. After this point every LDS dependency is
  // wave-local (each wave reads/writes only its own 16 activation rows; weight
  // tables are read-only). Hardware keeps DS ops in order per wave.
#pragma unroll 1
  for (int t = 0; t < TDIM; ++t) {
    const float tv = t * DT_F;
    const float sn = __sinf(tv), cn = __cosf(tv);
    const float* np = noise + ((size_t)t * NDIM + gBase) * HDIM;

    // prefetch this step's noise rows early (global_prefetch_b8); consumed ~5 GEMMs later
#pragma unroll
    for (int r = 0; r < 8; ++r)
      __builtin_prefetch(np + (size_t)(r0 + rHi + r) * HDIM + colB * 8, 0, 0);

    // G1: tanh(y @ g_w1' + gb1_eff)   (time cols folded into bias)
    zero_acc(acc);
    gemm_acc(S.actY, S.W[3], r0, lane, acc);
#pragma unroll
    for (int c = 0; c < 8; ++c)
#pragma unroll
      for (int r = 0; r < 8; ++r) {
        int col = c * 16 + colB;
        acc[c][r] = FAST_TANH(acc[c][r] + S.gb1[col] + sn * S.gts[col] + cn * S.gtc[col]);
      }
    store_act_bf(S.actH, r0, lane, acc);

    // G2 + sigmoid(dot(., g_w3) + g_b3) per row
    zero_acc(acc);
    gemm_acc(S.actH, S.W[4], r0, lane, acc);
#pragma unroll
    for (int c = 0; c < 8; ++c)
#pragma unroll
      for (int r = 0; r < 8; ++r) acc[c][r] = FAST_TANH(acc[c][r] + S.gb2[c * 16 + colB]);
    float grow[8];
#pragma unroll
    for (int r = 0; r < 8; ++r) {
      float p = 0.f;
#pragma unroll
      for (int c = 0; c < 8; ++c) p += acc[c][r] * S.gw3[c * 16 + colB];
      p = red16(p) + S.gb3s;
      grow[r] = FAST_RCP(1.0f + FAST_EXP(-p));
    }

    // F1
    zero_acc(acc);
    gemm_acc(S.actY, S.W[0], r0, lane, acc);
#pragma unroll
    for (int c = 0; c < 8; ++c)
#pragma unroll
      for (int r = 0; r < 8; ++r) {
        int col = c * 16 + colB;
        acc[c][r] = FAST_TANH(acc[c][r] + S.fb1[col] + sn * S.fts[col] + cn * S.ftc[col]);
      }
    store_act_bf(S.actH, r0, lane, acc);

    // F2
    zero_acc(acc);
    gemm_acc(S.actH, S.W[1], r0, lane, acc);
#pragma unroll
    for (int c = 0; c < 8; ++c)
#pragma unroll
      for (int r = 0; r < 8; ++r) acc[c][r] = FAST_TANH(acc[c][r] + S.fb2[c * 16 + colB]);
    store_act_bf(S.actH, r0, lane, acc);

    // F3 -> drift; Euler update y += drift*dt + diff*sqrt(dt)*z
    zero_acc(acc);
    gemm_acc(S.actH, S.W[2], r0, lane, acc);
#pragma unroll
    for (int c = 0; c < 8; ++c)
#pragma unroll
      for (int r = 0; r < 8; ++r) {
        int col = c * 16 + colB;
        int row = r0 + rHi + r;
        float z = np[(size_t)row * HDIM + col];
        y[c][r] += (acc[c][r] + S.fb3[col]) * DT_F + grow[r] * SQDT_F * z;
      }
    store_act_bf(S.actY, r0, lane, y);

    // decoder head: loc
    zero_acc(acc);
    gemm_acc(S.actY, S.W[5], r0, lane, acc);
#pragma unroll
    for (int c = 0; c < 8; ++c)
#pragma unroll
      for (int r = 0; r < 8; ++r) acc[c][r] += S.db1[c * 16 + colB];
    ln_relu(acc, colB, S.dgm, S.dbe);
#pragma unroll
    for (int r = 0; r < 8; ++r) {
      float p0 = 0.f, p1 = 0.f;
#pragma unroll
      for (int c = 0; c < 8; ++c) {
        int col = c * 16 + colB;
        p0 += acc[c][r] * S.dw2[col * 2 + 0];
        p1 += acc[c][r] * S.dw2[col * 2 + 1];
      }
      p0 = red16(p0) + S.db2s[0];
      p1 = red16(p1) + S.db2s[1];
      if (colB == 0) {
        size_t g = (size_t)(gBase + r0 + rHi + r);
        size_t idx = (g * TDIM + t) * 4;
        out_loc[idx + 0] = p0;
        out_loc[idx + 1] = p1;
      }
    }

    // scale head
    zero_acc(acc);
    gemm_acc(S.actY, S.W[6], r0, lane, acc);
#pragma unroll
    for (int c = 0; c < 8; ++c)
#pragma unroll
      for (int r = 0; r < 8; ++r) acc[c][r] += S.sb1[c * 16 + colB];
    ln_relu(acc, colB, S.sgm, S.sbe);
#pragma unroll
    for (int r = 0; r < 8; ++r) {
      float p0 = 0.f, p1 = 0.f;
#pragma unroll
      for (int c = 0; c < 8; ++c) {
        int col = c * 16 + colB;
        p0 += acc[c][r] * S.sw2[col * 2 + 0];
        p1 += acc[c][r] * S.sw2[col * 2 + 1];
      }
      p0 = red16(p0) + S.sb2s[0];
      p1 = red16(p1) + S.sb2s[1];
      if (colB == 0) {
        float e0 = (p0 > 0.f ? p0 : FAST_EXP(p0) - 1.0f) + 1.001f;
        float e1 = (p1 > 0.f ? p1 : FAST_EXP(p1) - 1.0f) + 1.001f;
        size_t g = (size_t)(gBase + r0 + rHi + r);
        size_t idx = (g * TDIM + t) * 4;
        out_loc[idx + 2] = e0;
        out_loc[idx + 3] = e1;
      }
    }
  }
}

// pi head: [N] rows, 16 rows/block, 16 lanes/row
__global__ __launch_bounds__(256) void pi_kernel(
    const float* __restrict__ local_embed, const float* __restrict__ global_embed,
    const float* __restrict__ p_w1, const float* __restrict__ p_b1,
    const float* __restrict__ p_g,  const float* __restrict__ p_be,
    const float* __restrict__ p_w2, const float* __restrict__ p_b2,
    float* __restrict__ out_pi) {
  const int tid = threadIdx.x;
  const int rloc = tid >> 4;
  const int sub = tid & 15;
  const size_t g = (size_t)blockIdx.x * 16 + rloc;
  const int m = (int)(g >> 12);
  const int a = (int)(g & (ADIM - 1));
  const float* le = local_embed + (size_t)a * HDIM;
  const float* ge = global_embed + g * HDIM;
  float h[8];
#pragma unroll
  for (int j = 0; j < 8; ++j) {
    int col = sub + j * 16;
    float s = p_b1[col];
    // pin = cat(local, global): rows 0..127 -> local, 128..255 -> global
    for (int k = 0; k < HDIM; ++k) s += le[k] * p_w1[(size_t)k * HDIM + col];
    for (int k = 0; k < HDIM; ++k) s += ge[k] * p_w1[(size_t)(k + HDIM) * HDIM + col];
    h[j] = s;
  }
  float sm = 0.f;
#pragma unroll
  for (int j = 0; j < 8; ++j) sm += h[j];
  float mu = red16(sm) * (1.0f / 128.0f);
  float sv = 0.f;
#pragma unroll
  for (int j = 0; j < 8; ++j) { float d = h[j] - mu; sv += d * d; }
  float rs = rsqrtf(red16(sv) * (1.0f / 128.0f) + EPSLN);
  float o = 0.f;
#pragma unroll
  for (int j = 0; j < 8; ++j) {
    int col = sub + j * 16;
    float v = (h[j] - mu) * rs * p_g[col] + p_be[col];
    v = v > 0.f ? v : 0.f;
    o += v * p_w2[col];
  }
  o = red16(o);
  if (sub == 0) out_pi[(size_t)a * MDIM + m] = o + p_b2[0];
}

__global__ void mask_kernel(const unsigned char* __restrict__ pm, float* __restrict__ out_mask) {
  int i = blockIdx.x * blockDim.x + threadIdx.x;
  if (i < ADIM * TDIM) {
    int a = i / TDIM, j = i - a * TDIM;
    out_mask[i] = pm[a * HIST + (HIST - TDIM) + j] ? 0.0f : 1.0f;
  }
}

extern "C" void kernel_launch(void* const* d_in, const int* in_sizes, int n_in,
                              void* d_out, int out_size, void* d_ws, size_t ws_size,
                              hipStream_t stream) {
  (void)in_sizes; (void)n_in; (void)d_ws; (void)ws_size; (void)out_size;
  auto F = [&](int i) { return (const float*)d_in[i]; };
  float* out = (float*)d_out;
  float* out_pi = out + (size_t)MDIM * ADIM * TDIM * 4;
  float* out_mask = out_pi + (size_t)ADIM * MDIM;

  sde_scan_kernel<<<NDIM / 128, 256, sizeof(ScanLds), stream>>>(
      F(0), F(1), F(2),
      F(3), F(4), F(5), F(6),
      F(7), F(8), F(9), F(10), F(11), F(12),
      F(13), F(14), F(15), F(16), F(17), F(18),
      F(19), F(20), F(21), F(22), F(23), F(24),
      F(25), F(26), F(27), F(28), F(29), F(30),
      out);

  pi_kernel<<<NDIM / 16, 256, 0, stream>>>(
      F(0), F(1), F(31), F(32), F(33), F(34), F(35), F(36), out_pi);

  mask_kernel<<<(ADIM * TDIM + 255) / 256, 256, 0, stream>>>(
      (const unsigned char*)d_in[37], out_mask);
}
